// DualAttention_9028021256256
// MI455X (gfx1250) — matrix operand order repaired and checked by reading, not compiled, not run on hardware
//
#include <hip/hip_runtime.h>

typedef __attribute__((ext_vector_type(16))) _Float16 v16h;
typedef __attribute__((ext_vector_type(8)))  float    v8f;
typedef __attribute__((ext_vector_type(4)))  float    f4;
typedef __attribute__((ext_vector_type(2)))  float    f2;

#define BB   64
#define LL   4096
#define DD   512
#define DPP  64
#define NCHUNK 16
#define CH   (LL / NCHUNK)          // 256 rows per chunk
#define NEG_INF (-__builtin_inff())

// ---------------------------------------------------------------------------
// WMMA fragment index helpers (CDNA5 16-bit layouts, wave32):
//  A (16xK tile, MxK): lane<16 -> M=lane, element i holds K = half*8+i (i<8)
//                      or 16+half*8+(i-8) (i>=8), half = lane>>4.
//  B (Kx16 tile, KxN): lane<16 -> N=lane, element i holds K as for A.
//  C/D (16x16 f32):    VGPR r -> M = half*8 + r, N = lane&15.
// ---------------------------------------------------------------------------
__device__ __forceinline__ int kidxA(int i, int half) {
    return (i < 8) ? (half * 8 + i) : (8 + half * 8 + i); // == 16+half*8+(i-8)
}
__device__ __forceinline__ int kidxB(int i, int half) {
    return (i < 8) ? (half * 8 + i) : (8 + half * 8 + i);
}

// ---------------------------------------------------------------------------
// Kernel 1: q_main[b,d] = sum_e source[b,e] * W_main_in[e,d]   (64x512, K=512)
//           q_pos [b,p] = sum_e source[b,e] * W_pos_in [e,p]   (64x64,  K=512)
// One wave per 16x16 output tile. blocks 0..127 -> q_main, 128..143 -> q_pos.
// ---------------------------------------------------------------------------
__global__ __launch_bounds__(32) void query_proj_kernel(
    const float* __restrict__ source,
    const float* __restrict__ Wmain,
    const float* __restrict__ Wpos,
    float* __restrict__ q_main,
    float* __restrict__ q_pos)
{
    int t = blockIdx.x;
    const float* Bw;
    float* out;
    int N, mt, nt;
    if (t < 128) { mt = t >> 5; nt = t & 31; Bw = Wmain; out = q_main; N = DD; }
    else { int u = t - 128; mt = u >> 2; nt = u & 3; Bw = Wpos; out = q_pos; N = DPP; }

    const int lane = threadIdx.x & 31;
    const int half = lane >> 4;
    const int mn   = lane & 15;
    const int arow = mt * 16 + mn;   // batch index for A rows
    const int bcol = nt * 16 + mn;   // output column for B

    v8f acc = {};
    for (int kk = 0; kk < DD; kk += 32) {
        v16h a, bf;
#pragma unroll
        for (int i = 0; i < 16; ++i) {
            int ka = kk + kidxA(i, half);
            int kb = kk + kidxB(i, half);
            a[i]  = (_Float16)source[arow * DD + ka];
            bf[i] = (_Float16)Bw[(size_t)kb * N + bcol];
        }
        acc = __builtin_amdgcn_wmma_f32_16x16x32_f16(
            false, a, false, bf, (short)0, acc, false, false);
    }
#pragma unroll
    for (int r = 0; r < 8; ++r) {
        int row = mt * 16 + half * 8 + r;
        out[row * N + bcol] = acc[r];
    }
}

// ---------------------------------------------------------------------------
// Kernel 2: fused streaming pass over main_embs (L,B,D) / pos_embs (L,B,DP).
// Grid (B, NCHUNK), 256 threads = 8 waves. Each wave processes one l-row per
// iteration: 2KB coalesced NT row load (read-once stream, keep L2 clean),
// dual dot product + wave allreduce, online softmax stats for ma / pa / av,
// and online-rescaled c_t accumulation while the row is still in registers
// -> a single HBM pass over the 512MB memory bank.
// ---------------------------------------------------------------------------
__global__ __launch_bounds__(256) void score_stream_kernel(
    const float* __restrict__ main_embs,   // (L,B,D) time-major
    const float* __restrict__ pos_embs,    // (L,B,DP)
    const float* __restrict__ q_main,
    const float* __restrict__ q_pos,
    const int*   __restrict__ memory_lengths,
    float* __restrict__ sc_main,           // (B,L) masked scores
    float* __restrict__ sc_pos,            // (B,L)
    float* __restrict__ stats,             // (B,NCHUNK,8): M1,S1,M2,S2,M3,S3
    float* __restrict__ ctpart)            // (B,NCHUNK,D)
{
    __shared__ float lds_ct[8 * DD];       // 16 KB: per-wave c_t partials
    __shared__ float lds_stats[8 * 6];
    __shared__ float lds_m3b[1];

    const int b     = blockIdx.x;
    const int chunk = blockIdx.y;
    const int lane  = threadIdx.x & 31;
    const int wave  = threadIdx.x >> 5;
    const int memlen = memory_lengths[b];

    // per-lane slices of the folded queries (d = s*128 + lane*4 + c)
    f4 qm[4];
#pragma unroll
    for (int s = 0; s < 4; ++s)
        qm[s] = *(const f4*)(q_main + b * DD + s * 128 + lane * 4);
    f2 qp = *(const f2*)(q_pos + b * DPP + lane * 2);

    float m1 = NEG_INF, s1 = 0.f, m2 = NEG_INF, s2 = 0.f, m3 = NEG_INF, s3 = 0.f;
    float ct[16];
#pragma unroll
    for (int j = 0; j < 16; ++j) ct[j] = 0.f;

    const int l0 = chunk * CH;
    for (int l = l0 + wave; l < l0 + CH; l += 8) {
        const float* mr = main_embs + ((size_t)l * BB + b) * DD;
        const float* pr = pos_embs  + ((size_t)l * BB + b) * DPP;
        // speculative prefetch of this wave's next row (global_prefetch_b8)
        __builtin_prefetch(mr + (size_t)8 * BB * DD + lane * 16, 0, 0);

        f4 v[4];
#pragma unroll
        for (int s = 0; s < 4; ++s)
            v[s] = __builtin_nontemporal_load(
                       (const f4*)(mr + s * 128 + lane * 4));
        f2 p = __builtin_nontemporal_load((const f2*)(pr + lane * 2));

        float dm = 0.f;
#pragma unroll
        for (int s = 0; s < 4; ++s)
            dm += v[s].x * qm[s].x + v[s].y * qm[s].y +
                  v[s].z * qm[s].z + v[s].w * qm[s].w;
        float dp = p.x * qp.x + p.y * qp.y;

#pragma unroll
        for (int off = 16; off > 0; off >>= 1) {
            dm += __shfl_xor(dm, off, 32);
            dp += __shfl_xor(dp, off, 32);
        }

        const bool valid = (l < memlen);           // wave-uniform
        if (lane == 0) {
            sc_main[(size_t)b * LL + l] = valid ? dm : NEG_INF;
            sc_pos [(size_t)b * LL + l] = valid ? dp : NEG_INF;
        }
        if (valid) {
            float nm1 = fmaxf(m1, dm);
            s1 = s1 * __expf(m1 - nm1) + __expf(dm - nm1); m1 = nm1;
            float nm2 = fmaxf(m2, dp);
            s2 = s2 * __expf(m2 - nm2) + __expf(dp - nm2); m2 = nm2;
            float sc  = dm + dp;
            float nm3 = fmaxf(m3, sc);
            float scale = __expf(m3 - nm3);
            float w     = __expf(sc - nm3);
            s3 = s3 * scale + w; m3 = nm3;
#pragma unroll
            for (int s = 0; s < 4; ++s) {
                ct[s * 4 + 0] = ct[s * 4 + 0] * scale + w * v[s].x;
                ct[s * 4 + 1] = ct[s * 4 + 1] * scale + w * v[s].y;
                ct[s * 4 + 2] = ct[s * 4 + 2] * scale + w * v[s].z;
                ct[s * 4 + 3] = ct[s * 4 + 3] * scale + w * v[s].w;
            }
        }
    }

    if (lane == 0) {
        float* st = lds_stats + wave * 6;
        st[0] = m1; st[1] = s1; st[2] = m2; st[3] = s2; st[4] = m3; st[5] = s3;
    }
    __syncthreads();

    if (threadIdx.x == 0) {
        float M[3] = {NEG_INF, NEG_INF, NEG_INF}, S[3] = {0.f, 0.f, 0.f};
        for (int w = 0; w < 8; ++w)
            for (int j = 0; j < 3; ++j)
                M[j] = fmaxf(M[j], lds_stats[w * 6 + 2 * j]);
        for (int w = 0; w < 8; ++w)
            for (int j = 0; j < 3; ++j) {
                float mw = lds_stats[w * 6 + 2 * j];
                if (mw > NEG_INF)
                    S[j] += lds_stats[w * 6 + 2 * j + 1] * __expf(mw - M[j]);
            }
        float* st = stats + ((size_t)b * NCHUNK + chunk) * 8;
        st[0] = M[0]; st[1] = S[0]; st[2] = M[1]; st[3] = S[1];
        st[4] = M[2]; st[5] = S[2];
        lds_m3b[0] = M[2];
    }
    __syncthreads();

    float M3b = lds_m3b[0];
    float cscale = (m3 > NEG_INF) ? __expf(m3 - M3b) : 0.f;
#pragma unroll
    for (int s = 0; s < 4; ++s)
#pragma unroll
        for (int c = 0; c < 4; ++c)
            lds_ct[wave * DD + s * 128 + lane * 4 + c] = ct[s * 4 + c] * cscale;
    __syncthreads();

    for (int d = threadIdx.x; d < DD; d += 256) {
        float acc = 0.f;
#pragma unroll
        for (int w = 0; w < 8; ++w) acc += lds_ct[w * DD + d];
        ctpart[((size_t)b * NCHUNK + chunk) * DD + d] = acc;
    }
}

// ---------------------------------------------------------------------------
// Kernel 3: per-batch reduce of chunk stats, assemble c_t, emit ma/pa/av.
// Outputs are written once and never re-read -> non-temporal stores.
// ---------------------------------------------------------------------------
__global__ __launch_bounds__(256) void finalize_kernel(
    const float* __restrict__ sc_main,
    const float* __restrict__ sc_pos,
    const float* __restrict__ stats,
    const float* __restrict__ ctpart,
    float* __restrict__ ct_out,            // ws (B,D)
    float* __restrict__ out_av,
    float* __restrict__ out_ma,
    float* __restrict__ out_pa)
{
    __shared__ float gs[6];
    const int b = blockIdx.x;

    if (threadIdx.x == 0) {
        float M[3] = {NEG_INF, NEG_INF, NEG_INF}, S[3] = {0.f, 0.f, 0.f};
        for (int c = 0; c < NCHUNK; ++c)
            for (int j = 0; j < 3; ++j)
                M[j] = fmaxf(M[j], stats[((size_t)b * NCHUNK + c) * 8 + 2 * j]);
        for (int c = 0; c < NCHUNK; ++c)
            for (int j = 0; j < 3; ++j) {
                float mc = stats[((size_t)b * NCHUNK + c) * 8 + 2 * j];
                if (mc > NEG_INF)
                    S[j] += stats[((size_t)b * NCHUNK + c) * 8 + 2 * j + 1] *
                            __expf(mc - M[j]);
            }
        gs[0] = M[0]; gs[1] = S[0]; gs[2] = M[1];
        gs[3] = S[1]; gs[4] = M[2]; gs[5] = S[2];
    }
    __syncthreads();
    const float M1 = gs[0], S1 = gs[1], M2 = gs[2], S2 = gs[3], M3 = gs[4], S3 = gs[5];
    const float r1 = 1.f / S1, r2 = 1.f / S2, r3 = 1.f / S3;

    for (int d = threadIdx.x; d < DD; d += 256) {
        float acc = 0.f;
        for (int c = 0; c < NCHUNK; ++c) {
            float mc = stats[((size_t)b * NCHUNK + c) * 8 + 4];
            if (mc > NEG_INF)
                acc += ctpart[((size_t)b * NCHUNK + c) * DD + d] * __expf(mc - M3);
        }
        ct_out[b * DD + d] = acc * r3;
    }

    for (int l = threadIdx.x; l < LL; l += 256) {
        float sm = sc_main[(size_t)b * LL + l];
        float sp = sc_pos [(size_t)b * LL + l];
        __builtin_nontemporal_store(__expf(sm - M1) * r1,      // 0 when masked
                                    out_ma + (size_t)b * LL + l);
        __builtin_nontemporal_store(__expf(sp - M2) * r2,
                                    out_pa + (size_t)b * LL + l);
        __builtin_nontemporal_store(__expf(sm + sp - M3) * r3, // softmax(main+pos)
                                    out_av + (size_t)b * LL + l);
    }
}

// ---------------------------------------------------------------------------
// Kernel 4: attn_h = tanh( [c_t, source] @ W_out^T )   (64x512, K=1024), WMMA.
// ---------------------------------------------------------------------------
__global__ __launch_bounds__(32) void out_proj_kernel(
    const float* __restrict__ ct,
    const float* __restrict__ source,
    const float* __restrict__ Wout,        // (D, 2D) row-major
    float* __restrict__ attn_h)
{
    const int t  = blockIdx.x;
    const int mt = t >> 5, nt = t & 31;    // 4 x 32 tiles of 16x16
    const int lane = threadIdx.x & 31;
    const int half = lane >> 4;
    const int mn   = lane & 15;
    const int arow = mt * 16 + mn;
    const int bcol = nt * 16 + mn;

    v8f acc = {};
    for (int kk = 0; kk < 2 * DD; kk += 32) {
        v16h a, bf;
#pragma unroll
        for (int i = 0; i < 16; ++i) {
            int ka = kk + kidxA(i, half);
            int kb = kk + kidxB(i, half);
            float av = (ka < DD) ? ct[arow * DD + ka]
                                 : source[arow * DD + (ka - DD)];
            a[i]  = (_Float16)av;
            bf[i] = (_Float16)Wout[(size_t)bcol * (2 * DD) + kb]; // B[k][n]=Wout[n][k]
        }
        acc = __builtin_amdgcn_wmma_f32_16x16x32_f16(
            false, a, false, bf, (short)0, acc, false, false);
    }
#pragma unroll
    for (int r = 0; r < 8; ++r) {
        int row = mt * 16 + half * 8 + r;
        attn_h[row * DD + bcol] = tanhf(acc[r]);
    }
}

// ---------------------------------------------------------------------------
extern "C" void kernel_launch(void* const* d_in, const int* in_sizes, int n_in,
                              void* d_out, int out_size, void* d_ws, size_t ws_size,
                              hipStream_t stream) {
    const float* source    = (const float*)d_in[0];
    const float* main_embs = (const float*)d_in[1];
    const float* pos_embs  = (const float*)d_in[2];
    const float* Wmain     = (const float*)d_in[3];
    const float* Wpos      = (const float*)d_in[4];
    const float* Wout      = (const float*)d_in[5];
    const int*   memlen    = (const int*)d_in[6];

    float* out     = (float*)d_out;          // attn_h, av, ma, pa (flat)
    float* attn_h  = out;
    float* out_av  = out + BB * DD;
    float* out_ma  = out_av + (size_t)BB * LL;
    float* out_pa  = out_ma + (size_t)BB * LL;

    float* ws      = (float*)d_ws;           // ~4.6 MB total
    float* q_main  = ws;                                  // B*D
    float* q_pos   = q_main + BB * DD;                    // B*DP
    float* sc_main = q_pos  + BB * DPP;                   // B*L
    float* sc_pos  = sc_main + (size_t)BB * LL;           // B*L
    float* stats   = sc_pos  + (size_t)BB * LL;           // B*NCHUNK*8
    float* ctpart  = stats   + BB * NCHUNK * 8;           // B*NCHUNK*D
    float* ct      = ctpart  + (size_t)BB * NCHUNK * DD;  // B*D

    query_proj_kernel<<<144, 32, 0, stream>>>(source, Wmain, Wpos, q_main, q_pos);
    score_stream_kernel<<<dim3(BB, NCHUNK), 256, 0, stream>>>(
        main_embs, pos_embs, q_main, q_pos, memlen,
        sc_main, sc_pos, stats, ctpart);
    finalize_kernel<<<BB, 256, 0, stream>>>(
        sc_main, sc_pos, stats, ctpart, ct, out_av, out_ma, out_pa);
    out_proj_kernel<<<128, 32, 0, stream>>>(ct, source, Wout, attn_h);
}
